// GNNModel_86457691668579
// MI455X (gfx1250) — matrix operand order repaired and checked by reading, not compiled, not run on hardware
//
#include <hip/hip_runtime.h>
#include <hip/hip_bf16.h>

// ---------- types for WMMA ----------
typedef __attribute__((ext_vector_type(16))) __bf16          v16bf;
typedef __attribute__((ext_vector_type(16))) unsigned short  v16u;
typedef __attribute__((ext_vector_type(8)))  float           v8f;

__device__ __forceinline__ unsigned short f2bf_bits(float f) {
    union { float f; unsigned u; } x; x.f = f;
    unsigned u = x.u;
    return (unsigned short)((u + 0x7FFFu + ((u >> 16) & 1u)) >> 16);  // RNE
}

// ---------- small utility kernels ----------
__global__ void zero_f32_kernel(float* __restrict__ p, long long n) {
    long long i = (long long)blockIdx.x * blockDim.x + threadIdx.x;
    long long stride = (long long)gridDim.x * blockDim.x;
    for (; i < n; i += stride) p[i] = 0.0f;
}

__global__ void degree_kernel(const int* __restrict__ dst, int E, float* __restrict__ deg) {
    long long i = (long long)blockIdx.x * blockDim.x + threadIdx.x;
    long long stride = (long long)gridDim.x * blockDim.x;
    for (; i < E; i += stride) atomicAdd(&deg[dst[i]], 1.0f);
}

__global__ void dinv_kernel(float* __restrict__ deg, int N) {
    int i = blockIdx.x * blockDim.x + threadIdx.x;
    if (i < N) {
        float d = deg[i] + 1.0f;          // self loop
        deg[i] = rsqrtf(d);               // deg >= 1 always
    }
}

// ---------- weight pre-pack: f32 [K x Mvalid] -> bf16 in WMMA B-fragment order ----------
// Fragment index = (colTile*(K/32) + kstep)*32 + lane; each lane owns 16 contiguous bf16
// (32 B, 32 B aligned). N=col0+(lane&15); t<8: K=k0+8*(lane>>4)+t ; t>=8: K=k0+16+8*(lane>>4)+(t-8).
// Columns >= Mvalid are zero-padded here, so the GEMM needs no predication.
__global__ void pack_w_kernel(const float* __restrict__ W, int ldw, int K, int Mvalid,
                              int nColTiles, unsigned short* __restrict__ Wp)
{
    int idx = blockIdx.x * blockDim.x + threadIdx.x;
    int ksteps = K >> 5;
    int total = nColTiles * ksteps * 32;
    if (idx >= total) return;
    int lane  = idx & 31;
    int rest  = idx >> 5;
    int kstep = rest % ksteps;
    int ct    = rest / ksteps;
    int half  = lane >> 4;
    int col   = ct * 16 + (lane & 15);
    int kb    = kstep * 32 + half * 8;
    unsigned short* outp = Wp + (size_t)idx * 16;
    for (int t = 0; t < 16; ++t) {
        float w = (col < Mvalid) ? W[(size_t)(kb + t + (t >= 8 ? 8 : 0)) * ldw + col] : 0.0f;
        outp[t] = f2bf_bits(w);
    }
}

// ---------- WMMA GEMM: C = act( maybe_relu(A[16r x K]) @ Wpacked + bias ) ----------
// One wave32 per 16x16 output tile. bf16 operands, f32 accumulate.
__global__ __launch_bounds__(128) void gemm_wmma_bf16(
    const float* __restrict__ A, int lda,
    const unsigned short* __restrict__ Wp,      // fragment-packed bf16 weights
    const float* __restrict__ bias,
    float* __restrict__ Cout, int ldc,
    int nRowTiles, int nColTiles, int K, int Mvalid,
    int reluIn, int reluOut)
{
    const int wavesPerBlock = blockDim.x >> 5;
    int wave = blockIdx.x * wavesPerBlock + (threadIdx.x >> 5);
    int totTiles = nRowTiles * nColTiles;
    if (wave >= totTiles) return;                  // wave-uniform: EXEC stays all-ones

    const int lane = threadIdx.x & 31;
    const int half = lane >> 4;                    // 0: lanes 0-15, 1: lanes 16-31
    const int m    = lane & 15;

    const int rowTile = wave / nColTiles;
    const int colTile = wave - rowTile * nColTiles;
    const int row0 = rowTile << 4;
    const int col0 = colTile << 4;
    const int col  = col0 + m;
    const int ksteps = K >> 5;

    v8f acc = {};
    const float* arow = A + (size_t)(row0 + m) * lda;
    const v16u* wfrag = (const v16u*)Wp + (size_t)colTile * ksteps * 32 + lane;

    for (int ks = 0; ks < ksteps; ++ks) {
        const int k0 = ks << 5;
        // --- B fragment: one aligned 32B vector load, no predication
        v16u bu = wfrag[(size_t)ks * 32];
        // --- A fragment: 16x32 bf16; lane holds row m, K = k0+half*8+{0..7} and +16
        const int ka = k0 + half * 8;
        float4 a0 = *(const float4*)(arow + ka);
        float4 a1 = *(const float4*)(arow + ka + 4);
        float4 a2 = *(const float4*)(arow + ka + 16);
        float4 a3 = *(const float4*)(arow + ka + 20);
        if (reluIn) {
            a0.x = fmaxf(a0.x, 0.f); a0.y = fmaxf(a0.y, 0.f); a0.z = fmaxf(a0.z, 0.f); a0.w = fmaxf(a0.w, 0.f);
            a1.x = fmaxf(a1.x, 0.f); a1.y = fmaxf(a1.y, 0.f); a1.z = fmaxf(a1.z, 0.f); a1.w = fmaxf(a1.w, 0.f);
            a2.x = fmaxf(a2.x, 0.f); a2.y = fmaxf(a2.y, 0.f); a2.z = fmaxf(a2.z, 0.f); a2.w = fmaxf(a2.w, 0.f);
            a3.x = fmaxf(a3.x, 0.f); a3.y = fmaxf(a3.y, 0.f); a3.z = fmaxf(a3.z, 0.f); a3.w = fmaxf(a3.w, 0.f);
        }
        v16u au;
        au[0]  = f2bf_bits(a0.x); au[1]  = f2bf_bits(a0.y); au[2]  = f2bf_bits(a0.z); au[3]  = f2bf_bits(a0.w);
        au[4]  = f2bf_bits(a1.x); au[5]  = f2bf_bits(a1.y); au[6]  = f2bf_bits(a1.z); au[7]  = f2bf_bits(a1.w);
        au[8]  = f2bf_bits(a2.x); au[9]  = f2bf_bits(a2.y); au[10] = f2bf_bits(a2.z); au[11] = f2bf_bits(a2.w);
        au[12] = f2bf_bits(a3.x); au[13] = f2bf_bits(a3.y); au[14] = f2bf_bits(a3.z); au[15] = f2bf_bits(a3.w);

        v16bf a = __builtin_bit_cast(v16bf, au);
        v16bf b = __builtin_bit_cast(v16bf, bu);
        acc = __builtin_amdgcn_wmma_f32_16x16x32_bf16(
                  false, a, false, b, (short)0, acc, false, false);
    }

    float bv = 0.0f;
    if (bias != nullptr && col < Mvalid) bv = bias[col];
    #pragma unroll
    for (int r = 0; r < 8; ++r) {
        int mr = r + half * 8;                     // C/D layout: VGPR r -> row r (+8 upper lanes)
        float v = acc[r] + bv;
        if (reluOut) v = fmaxf(v, 0.0f);
        Cout[(size_t)(row0 + mr) * ldc + col] = v; // pad cols (>=Mvalid) store exact 0
    }
}

// ---------- GCN aggregation ----------
// out[i][j] = bias_j + hw[i][j] * dinv[i]^2     (self-loop folded in, non-atomic full write)
__global__ void agg_init_kernel(const float* __restrict__ hw, const float* __restrict__ dinv,
                                const float* __restrict__ bias, float* __restrict__ out,
                                int N, int M, int Mvalid)
{
    long long idx = (long long)blockIdx.x * blockDim.x + threadIdx.x;
    long long total = (long long)N * M;
    long long stride = (long long)gridDim.x * blockDim.x;
    for (; idx < total; idx += stride) {
        int i = (int)(idx / M);
        int j = (int)(idx - (long long)i * M);
        float b = (j < Mvalid) ? bias[j] : 0.0f;
        float di = dinv[i];
        out[idx] = b + hw[idx] * di * di;
    }
}

// out[dst] += hw[src] * (dinv[src]*dinv[dst]); one thread per (edge, 4 features)
__global__ void agg_edges_kernel(const float* __restrict__ hw, const float* __restrict__ dinv,
                                 const int* __restrict__ src, const int* __restrict__ dst,
                                 int E, float* __restrict__ out, int M)
{
    const int vper = M >> 2;                       // float4 groups per row
    long long idx = (long long)blockIdx.x * blockDim.x + threadIdx.x;
    long long total = (long long)E * vper;
    long long stride = (long long)gridDim.x * blockDim.x;
    for (; idx < total; idx += stride) {
        int e = (int)(idx / vper);
        int q = (int)(idx - (long long)e * vper);
        int s = src[e];
        int d = dst[e];
        float w = dinv[s] * dinv[d];
        const float4 v = ((const float4*)(hw + (size_t)s * M))[q];
        float* o = out + (size_t)d * M + (size_t)q * 4;
        atomicAdd(o + 0, v.x * w);
        atomicAdd(o + 1, v.y * w);
        atomicAdd(o + 2, v.z * w);
        atomicAdd(o + 3, v.w * w);
    }
}

// ---------- pooling + log_softmax ----------
__global__ void pool_scatter_kernel(const float* __restrict__ h, const int* __restrict__ batch,
                                    int N, int ldh, int C,
                                    float* __restrict__ sums, float* __restrict__ cnt)
{
    int i = blockIdx.x * blockDim.x + threadIdx.x;
    if (i >= N) return;
    int g = batch[i];
    const float* row = h + (size_t)i * ldh;
    for (int c = 0; c < C; ++c) atomicAdd(&sums[(size_t)g * C + c], row[c]);
    atomicAdd(&cnt[g], 1.0f);
}

__global__ void pool_finish_kernel(const float* __restrict__ sums, const float* __restrict__ cnt,
                                   float* __restrict__ out, int G, int C)
{
    int g = blockIdx.x * blockDim.x + threadIdx.x;
    if (g >= G) return;
    float inv = 1.0f / fmaxf(cnt[g], 1.0f);
    float v[16];
    float mx = -3.4e38f;
    for (int c = 0; c < C; ++c) { v[c] = sums[(size_t)g * C + c] * inv; mx = fmaxf(mx, v[c]); }
    float se = 0.0f;
    for (int c = 0; c < C; ++c) se += expf(v[c] - mx);
    float lse = mx + logf(se);
    for (int c = 0; c < C; ++c) out[(size_t)g * C + c] = v[c] - lse;
}

// ---------- driver ----------
extern "C" void kernel_launch(void* const* d_in, const int* in_sizes, int n_in,
                              void* d_out, int out_size, void* d_ws, size_t ws_size,
                              hipStream_t stream) {
    const float* x      = (const float*)d_in[0];
    const int*   ei     = (const int*)d_in[1];
    const int*   batch  = (const int*)d_in[2];
    const float* emb_w1 = (const float*)d_in[3];
    const float* emb_b1 = (const float*)d_in[4];
    const float* emb_w2 = (const float*)d_in[5];
    const float* emb_b2 = (const float*)d_in[6];
    const float* w1 = (const float*)d_in[7];
    const float* b1 = (const float*)d_in[8];
    const float* w2 = (const float*)d_in[9];
    const float* b2 = (const float*)d_in[10];
    const float* w3 = (const float*)d_in[11];
    const float* b3 = (const float*)d_in[12];
    float* out = (float*)d_out;

    const int N  = in_sizes[2];            // 50000 (multiple of 16)
    const int D  = in_sizes[0] / N;        // 64
    const int E  = in_sizes[1] / 2;        // 800000
    const int H  = in_sizes[8];            // 128
    const int Cc = in_sizes[12];           // 10
    const int G  = out_size / Cc;          // 500
    const int Cpad = 16;

    const int* src = ei;
    const int* dst = ei + E;

    // workspace layout (f32): [dinv N][sums G*C][cnt G][pad8][hA N*H][hw N*H][hB N*H][Wp...]
    float* f0   = (float*)d_ws;
    float* dinv = f0;
    float* sums = f0 + N;
    float* cnt  = sums + (size_t)G * Cc;
    size_t off  = ((size_t)N + (size_t)G * Cc + (size_t)G + 7) & ~(size_t)7;   // 32B align
    float* hA = f0 + off;
    float* hw = hA + (size_t)N * H;
    float* hB = hw + (size_t)N * H;
    // fragment-packed weights (bf16), each block 32B aligned (sizes are multiples of 16 u16)
    unsigned short* wp_e1 = (unsigned short*)(hB + (size_t)N * H);
    size_t sz_e  = (size_t)(D / 16) * D * 16;      // colTiles*(K/32)*32*16 = colTiles*K*16
    unsigned short* wp_e2 = wp_e1 + sz_e;
    unsigned short* wp_w1 = wp_e2 + sz_e;
    size_t sz_w1 = (size_t)(H / 16) * D * 16;
    unsigned short* wp_w2 = wp_w1 + sz_w1;
    size_t sz_w2 = (size_t)(H / 16) * H * 16;
    unsigned short* wp_w3 = wp_w2 + sz_w2;
    (void)ws_size; (void)n_in;

    auto nblk = [](long long n, int t) { return (unsigned)((n + t - 1) / t); };

    const int nRowTiles = N / 16;

    // 0) pre-pack all weight matrices into WMMA B-fragment order (bf16, zero-padded)
    pack_w_kernel<<<nblk((D/16)*(D/32)*32, 128), 128, 0, stream>>>(emb_w1, D, D, D, D/16, wp_e1);
    pack_w_kernel<<<nblk((D/16)*(D/32)*32, 128), 128, 0, stream>>>(emb_w2, D, D, D, D/16, wp_e2);
    pack_w_kernel<<<nblk((H/16)*(D/32)*32, 128), 128, 0, stream>>>(w1,     H, D, H, H/16, wp_w1);
    pack_w_kernel<<<nblk((H/16)*(H/32)*32, 128), 128, 0, stream>>>(w2,     H, H, H, H/16, wp_w2);
    pack_w_kernel<<<nblk(1*(H/32)*32, 128),      128, 0, stream>>>(w3,     Cc, H, Cc, 1,  wp_w3);

    // 1) structure: zero accumulators, degree, dinv
    long long zn = (long long)N + (long long)G * Cc + G;
    zero_f32_kernel<<<nblk(zn, 256), 256, 0, stream>>>(f0, zn);
    degree_kernel<<<nblk(E, 256), 256, 0, stream>>>(dst, E, dinv);
    dinv_kernel<<<nblk(N, 256), 256, 0, stream>>>(dinv, N);

    // 2) MLP embedder: h = relu(x@W1+b1); h = relu(h@W2+b2)     [N,D]
    {
        int tiles = nRowTiles * (D / 16);
        gemm_wmma_bf16<<<nblk(tiles, 4), 128, 0, stream>>>(
            x, D, wp_e1, emb_b1, hA, D, nRowTiles, D / 16, D, D, 0, 1);
        gemm_wmma_bf16<<<nblk(tiles, 4), 128, 0, stream>>>(
            hA, D, wp_e2, emb_b2, hB, D, nRowTiles, D / 16, D, D, 0, 1);
    }

    // 3) GCN layer 1: hw = hB@w1 ; hA = b1 + norm-aggregate(hw)  [N,H]
    {
        int tiles = nRowTiles * (H / 16);
        gemm_wmma_bf16<<<nblk(tiles, 4), 128, 0, stream>>>(
            hB, D, wp_w1, nullptr, hw, H, nRowTiles, H / 16, D, H, 0, 0);
        agg_init_kernel<<<nblk((long long)N * H, 256), 256, 0, stream>>>(hw, dinv, b1, hA, N, H, H);
        agg_edges_kernel<<<nblk((long long)E * (H / 4), 256), 256, 0, stream>>>(hw, dinv, src, dst, E, hA, H);
    }

    // 4) GCN layer 2: hw = relu(hA)@w2 ; hB = b2 + aggregate(hw) [N,H]
    {
        int tiles = nRowTiles * (H / 16);
        gemm_wmma_bf16<<<nblk(tiles, 4), 128, 0, stream>>>(
            hA, H, wp_w2, nullptr, hw, H, nRowTiles, H / 16, H, H, 1, 0);
        agg_init_kernel<<<nblk((long long)N * H, 256), 256, 0, stream>>>(hw, dinv, b2, hB, N, H, H);
        agg_edges_kernel<<<nblk((long long)E * (H / 4), 256), 256, 0, stream>>>(hw, dinv, src, dst, E, hB, H);
    }

    // 5) GCN layer 3: hw = relu(hB)@w3 (cols padded to 16, pads exact 0) ; hA = b3 + aggregate
    {
        int tiles = nRowTiles * 1;
        gemm_wmma_bf16<<<nblk(tiles, 4), 128, 0, stream>>>(
            hB, H, wp_w3, nullptr, hw, Cpad, nRowTiles, 1, H, Cc, 1, 0);
        agg_init_kernel<<<nblk((long long)N * Cpad, 256), 256, 0, stream>>>(hw, dinv, b3, hA, N, Cpad, Cc);
        agg_edges_kernel<<<nblk((long long)E * (Cpad / 4), 256), 256, 0, stream>>>(hw, dinv, src, dst, E, hA, Cpad);
    }

    // 6) scatter-mean pool + log_softmax
    pool_scatter_kernel<<<nblk(N, 256), 256, 0, stream>>>(hA, batch, N, Cpad, Cc, sums, cnt);
    pool_finish_kernel<<<nblk(G, 128), 128, 0, stream>>>(sums, cnt, out, G, Cc);
}
